// GraphTransformerModule_33328946217388
// MI455X (gfx1250) — matrix operand order repaired and checked by reading, not compiled, not run on hardware
//
#include <hip/hip_runtime.h>
#include <math.h>

// ---------- CDNA5 WMMA types ----------
typedef __bf16 bf16_t;
typedef __attribute__((ext_vector_type(16))) __bf16 v16bf;
typedef __attribute__((ext_vector_type(8)))  __bf16 v8bf;
typedef __attribute__((ext_vector_type(4)))  __bf16 v4bf;
typedef __attribute__((ext_vector_type(8)))  float  v8f;

union V16U { v16bf v; v8bf h[2]; };

// order-preserving float<->uint map for atomicMax on floats
__device__ __forceinline__ unsigned ford(float f) {
  unsigned u = __float_as_uint(f);
  return (u & 0x80000000u) ? ~u : (u | 0x80000000u);
}
__device__ __forceinline__ float finvord(unsigned u) {
  return (u & 0x80000000u) ? __uint_as_float(u ^ 0x80000000u)
                           : __uint_as_float(~u);
}

// =====================================================================
// f32 -> bf16 conversion (optionally fused relu), 4 elems / thread
// =====================================================================
__global__ void f32_to_bf16(const float* __restrict__ in, bf16_t* __restrict__ out,
                            size_t n4, int doRelu)
{
  size_t t = (size_t)blockIdx.x * blockDim.x + threadIdx.x;
  if (t >= n4) return;
  float4 f = ((const float4*)in)[t];
  if (doRelu) {
    f.x = fmaxf(f.x, 0.f); f.y = fmaxf(f.y, 0.f);
    f.z = fmaxf(f.z, 0.f); f.w = fmaxf(f.w, 0.f);
  }
  v4bf o;
  o[0] = (bf16_t)f.x; o[1] = (bf16_t)f.y; o[2] = (bf16_t)f.z; o[3] = (bf16_t)f.w;
  ((v4bf*)out)[t] = o;
}

// =====================================================================
// WMMA GEMM: C[M,Nc] = A[M,K](bf16) @ B[K,Nc](f32) + bias, optional relu.
// One wave32 -> one 16x16 C tile via v_wmma_f32_16x16x32_bf16.
// Block = 4 waves = 64 M-rows, one 16-wide N panel.
// B panel staged in LDS *pre-permuted into WMMA B fragment order* so each
// lane reads its 32-byte fragment with wide ds loads.
// K is a template constant (32 or 128) -> fully unrolled wmma chain.
// =====================================================================
template <int K>
__global__ void __launch_bounds__(128)
gemm_wmma_bf16(const bf16_t* __restrict__ A, const float* __restrict__ B,
               const float* __restrict__ bias, float* __restrict__ C,
               int M, int Nc, int reluOut)
{
  __shared__ __align__(32) bf16_t Bs[K * 16];
  const int tid  = threadIdx.x;
  const int lane = tid & 31;
  const int wave = tid >> 5;
  const int ntile = blockIdx.y * 16;

  const int mrow = (blockIdx.x * 4 + wave) * 16;
  const int mA = mrow + (lane & 15);
  const int kA = (lane >> 4) * 16;

  // prefetch this wave's A rows while we stage B (global_prefetch_b8)
  if (mA < M) __builtin_prefetch(A + (size_t)mA * K, 0, 3);

  // stage B[:, ntile:ntile+16] into LDS in fragment order:
  //   dest elem = chunk*512 + lane*16 + i, where lane = (kin>>4)*16 + nn,
  //   i = kin&15, chunk = kk>>5, kin = kk&31
  for (int idx = tid; idx < K * 16; idx += 128) {
    int kk = idx >> 4, nn = idx & 15;
    int chunk = kk >> 5, kin = kk & 31;
    int dlane = ((kin >> 4) << 4) | nn;
    int i = kin & 15;
    Bs[chunk * 512 + dlane * 16 + i] = (bf16_t)B[(size_t)kk * Nc + ntile + nn];
  }
  __syncthreads();

  if (mrow >= M) return;

  const bf16_t* arow = A + (size_t)mA * K;
  const v16bf* bfrag = (const v16bf*)Bs;

  v8f acc = {};
#pragma unroll
  for (int k0 = 0; k0 < K; k0 += 32) {
    // A fragment (contiguous k = 16h + s, matching B staging): two 16B loads / lane
    V16U ua;
    ua.h[0] = *(const v8bf*)(arow + k0 + kA);
    ua.h[1] = *(const v8bf*)(arow + k0 + kA + 8);
    v16bf b = bfrag[(k0 >> 5) * 32 + lane];
    acc = __builtin_amdgcn_wmma_f32_16x16x32_bf16(
        /*neg_a=*/false, ua.v, /*neg_b=*/false, b,
        /*c_mod=*/(short)0, acc, /*reuse_a=*/false, /*reuse_b=*/false);
  }

  // C layout: VGPR r, lanes 0-15 -> row r, lanes 16-31 -> row 8+r; col = lane&15
  const int col = ntile + (lane & 15);
  const float bv = bias ? bias[col] : 0.f;
  const int rbase = mrow + ((lane >> 4) ? 8 : 0);
#pragma unroll
  for (int r = 0; r < 8; ++r) {
    float v = acc[r] + bv;
    if (reluOut) v = fmaxf(v, 0.f);
    C[(size_t)(rbase + r) * Nc + col] = v;
  }
}

// =====================================================================
// (optional relu) + LayerNorm over F features, one wave32 per row, in place
// =====================================================================
template <int F>
__global__ void relu_ln_kernel(float* __restrict__ h, const float* __restrict__ g,
                               const float* __restrict__ b, int M, int doRelu)
{
  constexpr int PER = F / 32;
  int row = blockIdx.x * (blockDim.x >> 5) + (threadIdx.x >> 5);
  int lane = threadIdx.x & 31;
  if (row >= M) return;
  float* p = h + (size_t)row * F;

  float v[PER];
  float s = 0.f;
#pragma unroll
  for (int i = 0; i < PER; ++i) {
    float x = p[lane * PER + i];
    if (doRelu) x = fmaxf(x, 0.f);
    v[i] = x; s += x;
  }
#pragma unroll
  for (int m = 16; m >= 1; m >>= 1) s += __shfl_xor(s, m, 32);
  float mu = s / (float)F;
  float vs = 0.f;
#pragma unroll
  for (int i = 0; i < PER; ++i) { float d = v[i] - mu; vs += d * d; }
#pragma unroll
  for (int m = 16; m >= 1; m >>= 1) vs += __shfl_xor(vs, m, 32);
  float inv = rsqrtf(vs / (float)F + 1e-5f);
#pragma unroll
  for (int i = 0; i < PER; ++i)
    p[lane * PER + i] = (v[i] - mu) * inv * g[lane * PER + i] + b[lane * PER + i];
}

// =====================================================================
// Edge attention passes (segment softmax + scatter), C = 32 floats
// =====================================================================
__global__ void edge_logits_max(const float* __restrict__ q, const float* __restrict__ k,
                                const int* __restrict__ src, const int* __restrict__ tgt,
                                float* __restrict__ alpha, unsigned* __restrict__ mbuf,
                                int E, int H, float scale)
{
  const int C4 = 8; // 32 floats = 8 float4
  int t = blockIdx.x * blockDim.x + threadIdx.x;
  if (t >= E * H) return;
  int e = t / H, h = t - e * H;
  int s = src[e], d = tgt[e];
  const float4* qp = (const float4*)(q + ((size_t)d * H + h) * 32);
  const float4* kp = (const float4*)(k + ((size_t)s * H + h) * 32);
  float acc = 0.f;
#pragma unroll
  for (int c = 0; c < C4; ++c) {
    float4 a = qp[c], b = kp[c];
    acc += a.x * b.x + a.y * b.y + a.z * b.z + a.w * b.w;
  }
  acc *= scale;
  alpha[t] = acc;
  atomicMax(&mbuf[(size_t)d * H + h], ford(acc));
}

__global__ void edge_exp_sum(const int* __restrict__ tgt, float* __restrict__ alpha,
                             const unsigned* __restrict__ mbuf, float* __restrict__ sbuf,
                             int E, int H)
{
  int t = blockIdx.x * blockDim.x + threadIdx.x;
  if (t >= E * H) return;
  int e = t / H, h = t - e * H;
  int d = tgt[e];
  float m = finvord(mbuf[(size_t)d * H + h]);
  float a = __expf(alpha[t] - m);
  alpha[t] = a;
  atomicAdd(&sbuf[(size_t)d * H + h], a);
}

__global__ void edge_scatter(const int* __restrict__ src, const int* __restrict__ tgt,
                             const float* __restrict__ alpha, const float* __restrict__ sbuf,
                             const float* __restrict__ v, float* __restrict__ out,
                             int E, int H)
{
  const int C4 = 8;
  int t = blockIdx.x * blockDim.x + threadIdx.x;
  if (t >= E * H) return;
  int e = t / H, h = t - e * H;
  int s = src[e], d = tgt[e];
  float w = alpha[t] / (sbuf[(size_t)d * H + h] + 1e-16f);
  const float4* vp = (const float4*)(v + ((size_t)s * H + h) * 32);
  float* op = out + ((size_t)d * H + h) * 32;
#pragma unroll
  for (int c = 0; c < C4; ++c) {
    float4 m4 = vp[c];
    atomicAdd(&op[c * 4 + 0], m4.x * w);
    atomicAdd(&op[c * 4 + 1], m4.y * w);
    atomicAdd(&op[c * 4 + 2], m4.z * w);
    atomicAdd(&op[c * 4 + 3], m4.w * w);
  }
}

// =====================================================================
// misc
// =====================================================================
__global__ void fill_u32(unsigned* __restrict__ p, unsigned v, size_t n)
{
  size_t t = (size_t)blockIdx.x * blockDim.x + threadIdx.x;
  if (t < n) p[t] = v;
}

__global__ void pool_accum(const float* __restrict__ y, const int* __restrict__ batch,
                           float* __restrict__ sums, float* __restrict__ cnt, int N, int Oc)
{
  int t = blockIdx.x * blockDim.x + threadIdx.x;
  if (t >= N * Oc) return;
  int n = t / Oc, c = t - n * Oc;
  int g = batch[n];
  atomicAdd(&sums[(size_t)g * Oc + c], y[(size_t)n * Oc + c]);
  if (c == 0) atomicAdd(&cnt[g], 1.0f);
}

__global__ void pool_final(const float* __restrict__ sums, const float* __restrict__ cnt,
                           float* __restrict__ out, int G, int Oc)
{
  int t = blockIdx.x * blockDim.x + threadIdx.x;
  if (t >= G * Oc) return;
  int g = t / Oc;
  out[t] = sums[t] / fmaxf(cnt[g], 1.0f);
}

// =====================================================================
// host side
// =====================================================================
static inline void launch_gemm(const bf16_t* A, const float* B, const float* bias,
                               float* C, int M, int K, int Nc, int reluOut,
                               hipStream_t s)
{
  dim3 grid((M + 63) / 64, Nc / 16);
  if (K == 32)
    gemm_wmma_bf16<32><<<grid, 128, 0, s>>>(A, B, bias, C, M, Nc, reluOut);
  else
    gemm_wmma_bf16<128><<<grid, 128, 0, s>>>(A, B, bias, C, M, Nc, reluOut);
}

extern "C" void kernel_launch(void* const* d_in, const int* in_sizes, int n_in,
                              void* d_out, int out_size, void* d_ws, size_t ws_size,
                              hipStream_t stream)
{
  (void)n_in; (void)ws_size;
  const float* x      = (const float*)d_in[0];
  const int*   ei     = (const int*)d_in[1];
  const int*   batch  = (const int*)d_in[2];
  const float* lin1_w = (const float*)d_in[3];
  const float* lin1_b = (const float*)d_in[4];
  const float* ln1_g  = (const float*)d_in[5];
  const float* ln1_bb = (const float*)d_in[6];
  const float* t1_wq  = (const float*)d_in[7];
  const float* t1_bq  = (const float*)d_in[8];
  const float* t1_wk  = (const float*)d_in[9];
  const float* t1_bk  = (const float*)d_in[10];
  const float* t1_wv  = (const float*)d_in[11];
  const float* t1_bv  = (const float*)d_in[12];
  const float* t1_ws  = (const float*)d_in[13];
  const float* t1_bs  = (const float*)d_in[14];
  const float* ln2_g  = (const float*)d_in[15];
  const float* ln2_bb = (const float*)d_in[16];
  const float* t2_wq  = (const float*)d_in[17];
  const float* t2_bq  = (const float*)d_in[18];
  const float* t2_wk  = (const float*)d_in[19];
  const float* t2_bk  = (const float*)d_in[20];
  const float* t2_wv  = (const float*)d_in[21];
  const float* t2_bv  = (const float*)d_in[22];
  const float* t2_ws  = (const float*)d_in[23];
  const float* t2_bs  = (const float*)d_in[24];
  const float* lin2_w = (const float*)d_in[25];
  const float* lin2_b = (const float*)d_in[26];

  const int IN_C = 128, HID = 32, H1 = 4, OUT_C = 64;
  const int N = in_sizes[2];
  const int E = in_sizes[1] / 2;
  const int G = out_size / OUT_C;
  const int* src = ei;
  const int* tgt = ei + E;

  // ---- workspace layout (float units; bf16 buffers = half float count) ----
  float* ws = (float*)d_ws;
  size_t off = 0;
  float* Hb    = ws + off; off += (size_t)N * HID;        // lin1 out / LN1
  float* Q1    = ws + off; off += (size_t)N * (H1 * HID); // also Q2 front
  float* K1    = ws + off; off += (size_t)N * (H1 * HID); // also K2 front
  float* V1    = ws + off; off += (size_t)N * (H1 * HID); // also V2 front
  float* O1    = ws + off; off += (size_t)N * (H1 * HID); // t1 out / h2
  float* ALPHA = ws + off; off += (size_t)E * H1;
  unsigned* MB = (unsigned*)(ws + off); off += (size_t)N * H1;
  float* SB    = ws + off; off += (size_t)N * H1;
  float* O2    = ws + off; off += (size_t)N * HID;        // t2 out
  float* Yb    = ws + off; off += (size_t)N * OUT_C;      // lin2 out
  float* PS    = ws + off; off += (size_t)G * OUT_C;      // pool sums
  float* PC    = ws + off; off += (size_t)G;              // pool counts
  bf16_t* Xb16  = (bf16_t*)(ws + off); off += (size_t)N * IN_C / 2;       // x in bf16
  bf16_t* Hb16  = (bf16_t*)(ws + off); off += (size_t)N * HID / 2;        // h1 in bf16
  bf16_t* H2b16 = (bf16_t*)(ws + off); off += (size_t)N * (H1 * HID) / 2; // h2 in bf16
  bf16_t* O2b16 = (bf16_t*)(ws + off); off += (size_t)N * HID / 2;        // relu(O2) bf16

  const float scale = 1.0f / sqrtf((float)HID);
  auto blocks = [](size_t n, int bs) { return (unsigned)((n + bs - 1) / bs); };

  // ---- lin1 + relu, then LN(32) ----
  f32_to_bf16<<<blocks((size_t)N * IN_C / 4, 256), 256, 0, stream>>>(x, Xb16, (size_t)N * IN_C / 4, 0);
  launch_gemm(Xb16, lin1_w, lin1_b, Hb, N, IN_C, HID, 1, stream);
  relu_ln_kernel<32><<<blocks(N, 8), 256, 0, stream>>>(Hb, ln1_g, ln1_bb, N, 0);
  f32_to_bf16<<<blocks((size_t)N * HID / 4, 256), 256, 0, stream>>>(Hb, Hb16, (size_t)N * HID / 4, 0);

  // ---- transformer1 projections (32 -> 128 each) ----
  launch_gemm(Hb16, t1_wq, t1_bq, Q1, N, HID, H1 * HID, 0, stream);
  launch_gemm(Hb16, t1_wk, t1_bk, K1, N, HID, H1 * HID, 0, stream);
  launch_gemm(Hb16, t1_wv, t1_bv, V1, N, HID, H1 * HID, 0, stream);
  launch_gemm(Hb16, t1_ws, t1_bs, O1, N, HID, H1 * HID, 0, stream); // skip into out

  // ---- transformer1 edge attention ----
  fill_u32<<<blocks((size_t)N * H1, 256), 256, 0, stream>>>(MB, 0u, (size_t)N * H1);
  fill_u32<<<blocks((size_t)N * H1, 256), 256, 0, stream>>>((unsigned*)SB, 0u, (size_t)N * H1);
  edge_logits_max<<<blocks((size_t)E * H1, 256), 256, 0, stream>>>(Q1, K1, src, tgt, ALPHA, MB, E, H1, scale);
  edge_exp_sum  <<<blocks((size_t)E * H1, 256), 256, 0, stream>>>(tgt, ALPHA, MB, SB, E, H1);
  edge_scatter  <<<blocks((size_t)E * H1, 256), 256, 0, stream>>>(src, tgt, ALPHA, SB, V1, O1, E, H1);

  // ---- relu + LN(128), then to bf16 ----
  relu_ln_kernel<128><<<blocks(N, 8), 256, 0, stream>>>(O1, ln2_g, ln2_bb, N, 1);
  f32_to_bf16<<<blocks((size_t)N * H1 * HID / 4, 256), 256, 0, stream>>>(O1, H2b16, (size_t)N * H1 * HID / 4, 0);

  // ---- transformer2 projections (128 -> 32 each), reuse Q1/K1/V1 fronts ----
  launch_gemm(H2b16, t2_wq, t2_bq, Q1, N, H1 * HID, HID, 0, stream);
  launch_gemm(H2b16, t2_wk, t2_bk, K1, N, H1 * HID, HID, 0, stream);
  launch_gemm(H2b16, t2_wv, t2_bv, V1, N, H1 * HID, HID, 0, stream);
  launch_gemm(H2b16, t2_ws, t2_bs, O2, N, H1 * HID, HID, 0, stream); // skip into out

  // ---- transformer2 edge attention (1 head) ----
  fill_u32<<<blocks(N, 256), 256, 0, stream>>>(MB, 0u, (size_t)N);
  fill_u32<<<blocks(N, 256), 256, 0, stream>>>((unsigned*)SB, 0u, (size_t)N);
  edge_logits_max<<<blocks(E, 256), 256, 0, stream>>>(Q1, K1, src, tgt, ALPHA, MB, E, 1, scale);
  edge_exp_sum  <<<blocks(E, 256), 256, 0, stream>>>(tgt, ALPHA, MB, SB, E, 1);
  edge_scatter  <<<blocks(E, 256), 256, 0, stream>>>(src, tgt, ALPHA, SB, V1, O2, E, 1);

  // ---- relu(O2) fused into bf16 convert, lin2: 32 -> 64 ----
  f32_to_bf16<<<blocks((size_t)N * HID / 4, 256), 256, 0, stream>>>(O2, O2b16, (size_t)N * HID / 4, 1);
  launch_gemm(O2b16, lin2_w, lin2_b, Yb, N, HID, OUT_C, 0, stream);

  // ---- global mean pool ----
  fill_u32<<<blocks((size_t)G * OUT_C, 256), 256, 0, stream>>>((unsigned*)PS, 0u, (size_t)G * OUT_C);
  fill_u32<<<blocks(G, 256), 256, 0, stream>>>((unsigned*)PC, 0u, (size_t)G);
  pool_accum<<<blocks((size_t)N * OUT_C, 256), 256, 0, stream>>>(Yb, batch, PS, PC, N, OUT_C);
  pool_final<<<blocks((size_t)G * OUT_C, 256), 256, 0, stream>>>(PS, PC, (float*)d_out, G, OUT_C);
}
